// PixNerfUpscaler_18743237280229
// MI455X (gfx1250) — matrix operand order repaired and checked by reading, not compiled, not run on hardware
//
#include <hip/hip_runtime.h>

// ---------------------------------------------------------------------------
// MI455X (gfx1250) implementation: bf16 WMMA everywhere, wave32, 320KB LDS.
// ---------------------------------------------------------------------------

typedef __attribute__((ext_vector_type(16))) __bf16 v16bf;
typedef __attribute__((ext_vector_type(8)))  float  v8f;

#define N_PATCH   8192      // B * (H/4) * (W/4)
#define C_CH      64
#define HID_DIM   256
#define TOKS      16

__device__ __forceinline__ float silu_f(float x) {
    return x / (1.0f + __expf(-x));
}

// --- WMMA fragment helpers (K-contiguous order chosen here for both A and B) ---------------
// 16-bit A matrix 16x32: lane m = lane&15, half = lane>>4.
//   K-contiguous: element s -> K = 16*half + s (matches B packing)
__device__ __forceinline__ v16bf load_fragA(const __bf16* __restrict__ base,
                                            int ld, int row, int k0, int half) {
    const __bf16* rp = base + row * ld + k0 + half * 16;
    v16bf f;
#pragma unroll
    for (int p4 = 0; p4 < 8; ++p4) {
        int kb = 2 * p4;
        f[2 * p4]     = rp[kb];
        f[2 * p4 + 1] = rp[kb + 1];
    }
    return f;
}

// 16-bit B matrix 32x16 (column per lane): lanes 0-15 hold K=0..15,
// lanes 16-31 hold K=16..31 -> each lane's 16 elems are K-contiguous = 32B.
__device__ __forceinline__ v16bf load_fragB32(const __bf16* __restrict__ p) {
    return *(const v16bf* __restrict__)p;
}

__device__ __forceinline__ v8f wmma_bf16(v16bf a, v16bf b, v8f c) {
    return __builtin_amdgcn_wmma_f32_16x16x32_bf16(
        false, a, false, b, (short)0, c, false, false);
}

// ---------------------------------------------------------------------------
// K0: unfold x[B,C,H,W] into proc[N, 16, C] (patch token rows)
// ---------------------------------------------------------------------------
__global__ void extract_patches(const float* __restrict__ x,
                                float* __restrict__ proc) {
    int idx = blockIdx.x * 256 + threadIdx.x;      // N*16*64 = 8388608 exact
    int c   = idx & 63;
    int tok = (idx >> 6) & 15;
    int n   = idx >> 10;
    int b   = n >> 10, gy = (n >> 5) & 31, gx = n & 31;
    int ty  = tok >> 2, tx = tok & 3;
    proc[idx] = x[((b * 64 + c) * 128 + gy * 4 + ty) * 128 + gx * 4 + tx];
}

// ---------------------------------------------------------------------------
// K1: patch-position MLP + summary[N,64]. One wave per patch, 2 channels/lane.
// ---------------------------------------------------------------------------
__global__ void prep_summary(const float* __restrict__ ppos,
                             const float* __restrict__ w1,
                             const float* __restrict__ b1,
                             const float* __restrict__ w2,
                             const float* __restrict__ b2,
                             float* __restrict__ summary) {
    int n    = (blockIdx.x * blockDim.x + threadIdx.x) >> 5;
    int lane = threadIdx.x & 31;
    int rem  = n & 1023;
    int gy   = rem >> 5, gx = rem & 31;
    float cy = (float)gy * (1.0f / 31.0f);
    float cx = (float)gx * (1.0f / 31.0f);
    int c0 = lane, c1 = lane + 32;
    float a0 = b2[c0], a1 = b2[c1];
#pragma unroll 4
    for (int k = 0; k < 256; ++k) {
        float h = silu_f(cy * w1[k] + cx * w1[256 + k] + b1[k]);
        a0 += h * w2[k * 64 + c0];
        a1 += h * w2[k * 64 + c1];
    }
    float m0 = 0.f, m1 = 0.f;
#pragma unroll
    for (int r = 0; r < 16; ++r) { m0 += ppos[r * 64 + c0]; m1 += ppos[r * 64 + c1]; }
    summary[n * 64 + c0] = a0 + m0 * 0.0625f;
    summary[n * 64 + c1] = a1 + m1 * 0.0625f;
}

// ---------------------------------------------------------------------------
// K2: pack a [K x NC] f32 weight into bf16 WMMA-B fragment-native order:
// dst[((ntile*ksteps + ks)*32 + lane)*16 + t] = src[(ks*32 + half*16 + t)*NC + col]
// ---------------------------------------------------------------------------
__global__ void pack_b(const float* __restrict__ src, __bf16* __restrict__ dst,
                       int K, int NC) {
    int tid    = blockIdx.x * blockDim.x + threadIdx.x;
    int ksteps = K >> 5;
    int total  = (NC >> 4) * ksteps * 32;
    if (tid >= total) return;
    int lane = tid & 31;
    int rem  = tid >> 5;
    int ks   = rem % ksteps;
    int ntl  = rem / ksteps;
    int col  = ntl * 16 + (lane & 15);
    int half = lane >> 4;
    __bf16* d = dst + (size_t)tid * 16;
#pragma unroll
    for (int t = 0; t < 16; ++t)
        d[t] = (__bf16)src[(ks * 32 + half * 16 + t) * NC + col];
}

// ---------------------------------------------------------------------------
// K3: per 16-patch tile, per block: hid = silu(summary@wn_w1+b), bhid likewise,
//     bias = bhid@bn_w2 + b2.  All via WMMA.  grid=(512, NB), 128 threads.
// ---------------------------------------------------------------------------
__global__ __launch_bounds__(128) void hid_bias_kernel(
    const float* __restrict__ summary,
    const __bf16* __restrict__ wn_w1p, const float* __restrict__ wn_b1,
    const __bf16* __restrict__ bn_w1p, const float* __restrict__ bn_b1,
    const __bf16* __restrict__ bn_w2p, const float* __restrict__ bn_b2,
    __bf16* __restrict__ hid_out, float* __restrict__ bias_out) {
    int blk = blockIdx.y;
    int pt  = blockIdx.x;
    __shared__ __bf16 As[16 * 64];
    __shared__ __bf16 Hs[16 * 256];
    __shared__ __bf16 Bs[16 * 256];
    int tid = threadIdx.x, lane = tid & 31, wave = tid >> 5;
    int half = lane >> 4, l16 = lane & 15;

    for (int i = tid; i < 1024; i += 128)
        As[i] = (__bf16)summary[pt * 1024 + i];
    __syncthreads();

    v16bf a0 = load_fragA(As, 64, l16, 0, half);
    v16bf a1 = load_fragA(As, 64, l16, 32, half);

    for (int which = 0; which < 2; ++which) {
        const __bf16* Wp = (which ? bn_w1p : wn_w1p) + (size_t)blk * 64 * 256;
        const float*  bb = (which ? bn_b1 : wn_b1) + blk * 256;
        __bf16* dstS = which ? Bs : Hs;
#pragma unroll
        for (int j = 0; j < 4; ++j) {
            int ntl = wave * 4 + j;
            v8f acc = {};
            acc = wmma_bf16(a0, load_fragB32(Wp + ((ntl * 2 + 0) * 32 + lane) * 16), acc);
            acc = wmma_bf16(a1, load_fragB32(Wp + ((ntl * 2 + 1) * 32 + lane) * 16), acc);
            int col = ntl * 16 + l16;
            float bc = bb[col];
#pragma unroll
            for (int r = 0; r < 8; ++r) {
                int m = r + half * 8;
                __bf16 v = (__bf16)silu_f(acc[r] + bc);
                dstS[m * 256 + col] = v;
                if (which == 0)
                    hid_out[((size_t)blk * N_PATCH + pt * 16 + m) * 256 + col] = v;
            }
        }
    }
    __syncthreads();

    // bias GEMM: [16,256] @ bn_w2[256,64]; 4 n-tiles, one per wave
    v8f acc = {};
    const __bf16* W2p = bn_w2p + (size_t)blk * 256 * 64;
#pragma unroll
    for (int ks = 0; ks < 8; ++ks) {
        v16bf af = load_fragA(Bs, 256, l16, ks * 32, half);
        acc = wmma_bf16(af, load_fragB32(W2p + ((wave * 8 + ks) * 32 + lane) * 16), acc);
    }
    int col = wave * 16 + l16;
    float bc = bn_b2[blk * 64 + col];
#pragma unroll
    for (int r = 0; r < 8; ++r) {
        int m = r + half * 8;
        bias_out[((size_t)blk * N_PATCH + pt * 16 + m) * 64 + col] = acc[r] + bc;
    }
}

// ---------------------------------------------------------------------------
// K4: fused hypernetwork block. One WG = 16 patches, 256 threads (8 waves).
//   Phase A: layernorm(proc) -> normed bf16 LDS
//   Phase B: wgt[16x4096] = hid @ wn_w2 (WMMA, 2048 wmma/WG) -> 128KB LDS
//   Phase C: per patch: proc += normed @ wgt^T + bias (WMMA)
// Dynamic LDS = 168KB -- relies on CDNA5's 320KB-per-WGP LDS.
// ---------------------------------------------------------------------------
__global__ __launch_bounds__(256) void block_apply(
    const float* __restrict__ ln_g, const float* __restrict__ ln_b,
    const __bf16* __restrict__ hid, const __bf16* __restrict__ wn_w2p,
    const float* __restrict__ wn_b2, const float* __restrict__ bias,
    float* __restrict__ proc, int blk) {
    extern __shared__ char smem[];
    __bf16* wgt = (__bf16*)smem;                       // 16 * 4096  (128KB)
    __bf16* nrm = (__bf16*)(smem + 131072);            // 16 * 16 * 64 (32KB)
    __bf16* Hs  = (__bf16*)(smem + 131072 + 32768);    // 16 * 256     (8KB)

    int pt = blockIdx.x;
    int tid = threadIdx.x, lane = tid & 31, wave = tid >> 5;
    int half = lane >> 4, l16 = lane & 15;

    // stage hid tile (bf16, row-major [16][256])
    for (int i = tid; i < 4096; i += 256)
        Hs[i] = hid[((size_t)blk * N_PATCH + pt * 16) * 256 + i];

    // layernorm: one (patch, token) row per thread
    {
        int p = tid >> 4, tok = tid & 15;
        const float* pr = proc + (((size_t)pt * 16 + p) * 16 + tok) * 64;
        float mean = 0.f;
#pragma unroll 8
        for (int c = 0; c < 64; ++c) mean += pr[c];
        mean *= (1.0f / 64.0f);
        float var = 0.f;
#pragma unroll 8
        for (int c = 0; c < 64; ++c) { float d = pr[c] - mean; var += d * d; }
        var *= (1.0f / 64.0f);
        float rstd = rsqrtf(var + 1e-5f);
        __bf16* nr = nrm + (p * 16 + tok) * 64;
#pragma unroll 8
        for (int c = 0; c < 64; ++c)
            nr[c] = (__bf16)((pr[c] - mean) * rstd * ln_g[blk * 64 + c] + ln_b[blk * 64 + c]);
    }
    __syncthreads();

    // A fragments for the hypernet GEMM (K=256 -> 8 fragments, reused 32x)
    v16bf afr[8];
#pragma unroll
    for (int ks = 0; ks < 8; ++ks)
        afr[ks] = load_fragA(Hs, 256, l16, ks * 32, half);

    // GEMM1: [16,256] @ wn_w2[256,4096] -> wgt LDS. 256 n-tiles / 8 waves.
    const __bf16* Wp = wn_w2p + (size_t)blk * 256 * 4096;
    for (int j = 0; j < 32; ++j) {
        int ntl = wave * 32 + j;
        if (j + 1 < 32)  // stream next tile's weights (global_prefetch_b8)
            __builtin_prefetch(Wp + ((size_t)(ntl + 1) * 8 * 32 + lane) * 16, 0, 1);
        v8f acc = {};
#pragma unroll
        for (int ks = 0; ks < 8; ++ks)
            acc = wmma_bf16(afr[ks],
                            load_fragB32(Wp + (((size_t)ntl * 8 + ks) * 32 + lane) * 16),
                            acc);
        int col = ntl * 16 + l16;                 // col = i*64 + j
        float b2c = wn_b2[blk * 4096 + col];
#pragma unroll
        for (int r = 0; r < 8; ++r) {
            int m = r + half * 8;                 // patch within tile
            wgt[m * 4096 + col] = (__bf16)(acc[r] + b2c);
        }
    }
    __syncthreads();

    // GEMM2 (apply): per patch  proc[p] += normed[p] @ wgt[p]^T + bias
    // 64 jobs (patch x 4 i-tiles) / 8 waves
#pragma unroll
    for (int j = 0; j < 8; ++j) {
        int job = wave * 8 + j;
        int pp = job >> 2, it = job & 3;
        const __bf16* An = nrm + pp * 1024;       // [16 tok][64 j]
        const __bf16* Bw = wgt + pp * 4096;       // [64 i][64 j] -> B is K-contig
        v8f acc = {};
#pragma unroll
        for (int ks = 0; ks < 2; ++ks) {
            v16bf af = load_fragA(An, 64, l16, ks * 32, half);
            v16bf bf = load_fragB32(Bw + (it * 16 + l16) * 64 + ks * 32 + half * 16);
            acc = wmma_bf16(af, bf, acc);
        }
        int ic = it * 16 + l16;
        int patch = pt * 16 + pp;
        float bc = bias[((size_t)blk * N_PATCH + patch) * 64 + ic];
#pragma unroll
        for (int r = 0; r < 8; ++r) {
            int tok = r + half * 8;
            float* dp = proc + (((size_t)patch * 16 + tok) * 64) + ic;
            *dp = acc[r] + bc + *dp;
        }
    }
}

// ---------------------------------------------------------------------------
// K5: 1x1 conv (64->256 via WMMA) + pixel shuffle r=2, fused scatter store.
// One WG = 16 consecutive w-pixels at fixed (b,h). grid = 8*128*8 = 8192.
// ---------------------------------------------------------------------------
__global__ __launch_bounds__(128) void conv_shuffle(
    const float* __restrict__ proc, const __bf16* __restrict__ convp,
    const float* __restrict__ conv_b, float* __restrict__ out) {
    int bid = blockIdx.x;
    int wt = bid & 7, h = (bid >> 3) & 127, b = bid >> 10;
    __shared__ __bf16 As[16 * 64];
    int tid = threadIdx.x, lane = tid & 31, wave = tid >> 5;
    int half = lane >> 4, l16 = lane & 15;

    for (int i = tid; i < 1024; i += 128) {
        int row = i >> 6, c = i & 63;
        int w = wt * 16 + row;
        int n = b * 1024 + (h >> 2) * 32 + (w >> 2);
        int tok = (h & 3) * 4 + (w & 3);
        As[i] = (__bf16)proc[((size_t)n * 16 + tok) * 64 + c];
    }
    __syncthreads();

    v16bf a0 = load_fragA(As, 64, l16, 0, half);
    v16bf a1 = load_fragA(As, 64, l16, 32, half);
#pragma unroll
    for (int j = 0; j < 4; ++j) {
        int ntl = wave * 4 + j;
        v8f acc = {};
        acc = wmma_bf16(a0, load_fragB32(convp + ((ntl * 2 + 0) * 32 + lane) * 16), acc);
        acc = wmma_bf16(a1, load_fragB32(convp + ((ntl * 2 + 1) * 32 + lane) * 16), acc);
        int o = ntl * 16 + l16;
        float cb = conv_b[o];
        int cout = o >> 2, r1 = (o >> 1) & 1, r2 = o & 1;
#pragma unroll
        for (int r = 0; r < 8; ++r) {
            int w = wt * 16 + r + half * 8;
            out[(((size_t)b * 64 + cout) * 256 + (2 * h + r1)) * 256 + (2 * w + r2)]
                = acc[r] + cb;
        }
    }
}

// ---------------------------------------------------------------------------
extern "C" void kernel_launch(void* const* d_in, const int* in_sizes, int n_in,
                              void* d_out, int out_size, void* d_ws, size_t ws_size,
                              hipStream_t stream) {
    (void)in_sizes; (void)n_in; (void)out_size; (void)ws_size;
    const float* x        = (const float*)d_in[0];
    const float* ppos     = (const float*)d_in[1];
    const float* ppe_w1   = (const float*)d_in[2];
    const float* ppe_b1   = (const float*)d_in[3];
    const float* ppe_w2   = (const float*)d_in[4];
    const float* ppe_b2   = (const float*)d_in[5];
    const float* ln_g     = (const float*)d_in[6];
    const float* ln_b     = (const float*)d_in[7];
    const float* wn_w1    = (const float*)d_in[8];
    const float* wn_b1    = (const float*)d_in[9];
    const float* wn_w2    = (const float*)d_in[10];
    const float* wn_b2    = (const float*)d_in[11];
    const float* bn_w1    = (const float*)d_in[12];
    const float* bn_b1    = (const float*)d_in[13];
    const float* bn_w2    = (const float*)d_in[14];
    const float* bn_b2    = (const float*)d_in[15];
    const float* conv_w   = (const float*)d_in[16];
    const float* conv_b   = (const float*)d_in[17];
    float* out = (float*)d_out;

    // workspace carve (~53 MB total)
    char* ws = (char*)d_ws;
    size_t off = 0;
    auto carve = [&](size_t bytes) -> void* {
        void* p = ws + off;
        off = (off + bytes + 255) & ~(size_t)255;
        return p;
    };
    float*  proc    = (float*)carve((size_t)N_PATCH * 16 * 64 * 4);   // 32MB
    float*  summary = (float*)carve((size_t)N_PATCH * 64 * 4);        //  2MB
    __bf16* hid     = (__bf16*)carve((size_t)2 * N_PATCH * 256 * 2);  //  8MB
    float*  bias    = (float*)carve((size_t)2 * N_PATCH * 64 * 4);    //  4MB
    __bf16* wn_w1p  = (__bf16*)carve((size_t)2 * 64 * 256 * 2);
    __bf16* wn_w2p  = (__bf16*)carve((size_t)2 * 256 * 4096 * 2);     //  4MB
    __bf16* bn_w1p  = (__bf16*)carve((size_t)2 * 64 * 256 * 2);
    __bf16* bn_w2p  = (__bf16*)carve((size_t)2 * 256 * 64 * 2);
    __bf16* convp   = (__bf16*)carve((size_t)64 * 256 * 2);

    // K0: unfold
    extract_patches<<<(N_PATCH * 16 * 64) / 256, 256, 0, stream>>>(x, proc);
    // K1: summary
    prep_summary<<<N_PATCH * 32 / 256, 256, 0, stream>>>(ppos, ppe_w1, ppe_b1,
                                                         ppe_w2, ppe_b2, summary);
    // K2: pack weights to WMMA-B fragment layout (bf16)
    for (int blk = 0; blk < 2; ++blk) {
        pack_b<<<4, 256, 0, stream>>>(wn_w1 + blk * 16384, wn_w1p + blk * 16384, 64, 256);
        pack_b<<<256, 256, 0, stream>>>(wn_w2 + (size_t)blk * 1048576,
                                        wn_w2p + (size_t)blk * 1048576, 256, 4096);
        pack_b<<<4, 256, 0, stream>>>(bn_w1 + blk * 16384, bn_w1p + blk * 16384, 64, 256);
        pack_b<<<4, 256, 0, stream>>>(bn_w2 + blk * 16384, bn_w2p + blk * 16384, 256, 64);
    }
    pack_b<<<4, 256, 0, stream>>>(conv_w, convp, 64, 256);

    // K3: hid / bias for both blocks
    hid_bias_kernel<<<dim3(N_PATCH / 16, 2), 128, 0, stream>>>(
        summary, wn_w1p, wn_b1, bn_w1p, bn_b1, bn_w2p, bn_b2, hid, bias);

    // K4: two sequential hypernetwork blocks (168KB dynamic LDS each)
    const size_t smem_bytes = 131072 + 32768 + 8192;
    block_apply<<<N_PATCH / 16, 256, smem_bytes, stream>>>(
        ln_g, ln_b, hid, wn_w2p, wn_b2, bias, proc, 0);
    block_apply<<<N_PATCH / 16, 256, smem_bytes, stream>>>(
        ln_g, ln_b, hid, wn_w2p, wn_b2, bias, proc, 1);

    // K5: conv + pixel shuffle
    conv_shuffle<<<8 * 128 * 8, 128, 0, stream>>>(proc, convp, conv_b, out);
}
